// AttentionDecoder_747324310026
// MI455X (gfx1250) — compile-verified, matrix operand order checked by reading, not run on hardware
//
#include <hip/hip_runtime.h>
#include <hip/hip_bf16.h>
#include <math.h>

// ---------------- problem constants (from reference) ----------------
#define BB      256
#define SS      256
#define HH      512
#define VV      128
#define MAXLEN  300
#define NPAD    304          // att logits N padded to 16
#define SOS_TOK 1
#define LDOUT   (MAXLEN * VV)   // 38400, row stride of d_out logits

typedef __bf16 bf16_t;
typedef bf16_t v8bf  __attribute__((ext_vector_type(8)));
typedef bf16_t v16bf __attribute__((ext_vector_type(16)));
typedef float  v8f_t __attribute__((ext_vector_type(8)));

__device__ __forceinline__ float sigf(float x) { return 1.0f / (1.0f + expf(-x)); }

// =====================================================================
// Generic bf16 WMMA GEMM:  C[M,N] = A[M,K] * Bt[N,K]^T + bias[N]
// One wave per 16x16 C tile, 8 waves per block, f32 accumulation via
// v_wmma_f32_16x16x32_bf16.  Optional relu, f32 store, bf16 store.
// =====================================================================
template <bool RELU, bool STORE_F32, bool STORE_BF16>
__global__ __launch_bounds__(256) void gemm_bf16_wmma(
    const bf16_t* __restrict__ A, const bf16_t* __restrict__ Bt,
    const float* __restrict__ bias,
    float* __restrict__ C, int ldc,
    bf16_t* __restrict__ Cbf, int ldcbf, int bfColOff,
    int Nreal, int Ntiles, int K)
{
    const int wave = threadIdx.x >> 5;
    const int lane = threadIdx.x & 31;
    const int tile = blockIdx.y * 8 + wave;
    if (tile >= Ntiles) return;                 // whole wave exits -> EXEC stays all-1s
    const int n0  = tile * 16;
    const int m0  = blockIdx.x * 16;
    const int mn  = lane & 15;                  // A row / B column within tile
    const int sel = lane >> 4;                  // K-half selector (ISA 7.12.2)

    const bf16_t* ap = A  + (size_t)(m0 + mn) * K + sel * 8;
    const bf16_t* bp = Bt + (size_t)(n0 + mn) * K + sel * 8;

    v8f_t acc = {};
    for (int k = 0; k < K; k += 32) {
        v8bf alo = *(const v8bf*)(ap + k);
        v8bf ahi = *(const v8bf*)(ap + k + 16);
        v8bf blo = *(const v8bf*)(bp + k);
        v8bf bhi = *(const v8bf*)(bp + k + 16);
        v16bf af = __builtin_shufflevector(alo, ahi, 0,1,2,3,4,5,6,7,8,9,10,11,12,13,14,15);
        v16bf bf = __builtin_shufflevector(blo, bhi, 0,1,2,3,4,5,6,7,8,9,10,11,12,13,14,15);
        acc = __builtin_amdgcn_wmma_f32_16x16x32_bf16(
                  /*neg_a=*/false, af, /*neg_b=*/false, bf,
                  /*c_mod=*/(short)0, acc, /*reuse_a=*/false, /*reuse_b=*/false);
    }

    const int col = n0 + mn;
    if (col < Nreal) {
        const float bv = bias[col];
#pragma unroll
        for (int j = 0; j < 8; ++j) {
            const int row = m0 + j + 8 * sel;   // C/D layout: VGPR j, lane-half sel
            float v = acc[j] + bv;
            if (RELU) v = v > 0.0f ? v : 0.0f;
            if (STORE_F32)  C[(size_t)row * ldc + col] = v;
            if (STORE_BF16) Cbf[(size_t)row * ldcbf + bfColOff + col] = (bf16_t)v;
        }
    }
}

// ---------------- one-time setup kernels ----------------
__global__ void k_cast_bf16(const float* __restrict__ src, bf16_t* __restrict__ dst, int n)
{
    int i = blockIdx.x * blockDim.x + threadIdx.x;
    if (i < n) dst[i] = (bf16_t)src[i];
}

__global__ void k_build_attW(const float* __restrict__ attW, bf16_t* __restrict__ dst)
{   // [NPAD, 2H], rows >= MAXLEN zero
    int i = blockIdx.x * blockDim.x + threadIdx.x;
    if (i >= NPAD * 2 * HH) return;
    int row = i >> 10, col = i & 1023;
    dst[i] = (row < MAXLEN) ? (bf16_t)attW[row * (2 * HH) + col] : (bf16_t)0.0f;
}

__global__ void k_build_Wg(const float* __restrict__ Wih, const float* __restrict__ Whh,
                           bf16_t* __restrict__ dst)
{   // [4H, 2H] = [W_ih | W_hh]
    int i = blockIdx.x * blockDim.x + threadIdx.x;
    if (i >= 4 * HH * 2 * HH) return;
    int row = i >> 10, col = i & 1023;
    float v = (col < HH) ? Wih[row * HH + col] : Whh[row * HH + (col - HH)];
    dst[i] = (bf16_t)v;
}

__global__ void k_build_bsum(const float* __restrict__ bih, const float* __restrict__ bhh,
                             float* __restrict__ dst)
{
    int i = blockIdx.x * blockDim.x + threadIdx.x;
    if (i < 4 * HH) dst[i] = bih[i] + bhh[i];
}

__global__ void k_build_encoded(const int* __restrict__ enc_out, const float* __restrict__ emb,
                                bf16_t* __restrict__ enc_bf)
{   // encoded[b,s,h] = emb[enc_out[b,s],h] + pe[S][h]   (single PE row, per reference)
    int i = blockIdx.x * blockDim.x + threadIdx.x;     // over B*S*H
    if (i >= BB * SS * HH) return;
    int h = i & (HH - 1);
    int bs = i >> 9;
    int tok = enc_out[bs];
    float arg = 256.0f * expf((float)(2 * (h >> 1)) * (-logf(10000.0f) / (float)HH));
    float pe  = (h & 1) ? cosf(arg) : sinf(arg);
    enc_bf[i] = (bf16_t)(emb[tok * HH + h] + pe);
}

__global__ void k_init_state(const bf16_t* __restrict__ emb_bf,
                             float* __restrict__ c, bf16_t* __restrict__ h_bf,
                             bf16_t* __restrict__ att_in, bf16_t* __restrict__ comb_in,
                             bf16_t* __restrict__ xh_g, float* __restrict__ loss_acc)
{
    int i = blockIdx.x * blockDim.x + threadIdx.x;     // over B*H
    if (i >= BB * HH) return;
    int b = i >> 9, h = i & (HH - 1);
    c[i] = 0.0f;
    h_bf[i] = (bf16_t)0.0f;
    att_in[b * 1024 + HH + h] = (bf16_t)0.0f;
    xh_g [b * 1024 + HH + h] = (bf16_t)0.0f;
    bf16_t e = emb_bf[SOS_TOK * HH + h];
    att_in [b * 1024 + h] = e;
    comb_in[b * 1024 + h] = e;
    if (i == 0) *loss_acc = 0.0f;
}

// ---------------- per-step kernels ----------------
__global__ __launch_bounds__(256) void k_softmax_att(const float* __restrict__ logits,
                                                     float* __restrict__ att_wf)
{   // one block per batch row; softmax over 300 cols; store first 256 probs (rest hit zero pad)
    __shared__ float red[256];
    int b = blockIdx.x, t = threadIdx.x;
    float x1 = logits[b * NPAD + t];
    float x2 = (t < MAXLEN - 256) ? logits[b * NPAD + 256 + t] : -3.0e38f;
    float mx = fmaxf(x1, x2);
    red[t] = mx; __syncthreads();
    for (int s = 128; s > 0; s >>= 1) { if (t < s) red[t] = fmaxf(red[t], red[t + s]); __syncthreads(); }
    mx = red[0]; __syncthreads();
    float e1 = expf(x1 - mx);
    float e2 = (t < MAXLEN - 256) ? expf(x2 - mx) : 0.0f;
    red[t] = e1 + e2; __syncthreads();
    for (int s = 128; s > 0; s >>= 1) { if (t < s) red[t] += red[t + s]; __syncthreads(); }
    att_wf[b * 256 + t] = e1 / red[0];
}

__global__ __launch_bounds__(256) void k_context(const float* __restrict__ att_wf,
                                                 const bf16_t* __restrict__ enc_bf,
                                                 bf16_t* __restrict__ comb_in)
{   // a[b,h] = sum_{l<256} w[b,l]*enc[b,l,h]  -> comb_in[b, 512+h]
    __shared__ float w[256];
    int b = blockIdx.x, t = threadIdx.x;
    w[t] = att_wf[b * 256 + t];
    __syncthreads();
    const bf16_t* e = enc_bf + (size_t)b * SS * HH;
    float a0 = 0.0f, a1 = 0.0f;
    for (int l = 0; l < SS; ++l) {
        float wl = w[l];
        a0 += wl * (float)e[l * HH + t];
        a1 += wl * (float)e[l * HH + t + 256];
    }
    comb_in[b * 1024 + HH + t]       = (bf16_t)a0;
    comb_in[b * 1024 + HH + t + 256] = (bf16_t)a1;
}

__global__ __launch_bounds__(256) void k_lstm_ew(const float* __restrict__ gates,
                                                 float* __restrict__ c,
                                                 bf16_t* __restrict__ h_bf,
                                                 bf16_t* __restrict__ att_in,
                                                 bf16_t* __restrict__ xh_g)
{
    int i = blockIdx.x * blockDim.x + threadIdx.x;   // over B*H
    if (i >= BB * HH) return;
    int b = i >> 9, h = i & (HH - 1);
    const float* g = gates + (size_t)b * 4 * HH;
    float gi = g[h], gf = g[HH + h], gg = g[2 * HH + h], go = g[3 * HH + h];
    float cn = sigf(gf) * c[i] + sigf(gi) * tanhf(gg);
    float hn = sigf(go) * tanhf(cn);
    c[i] = cn;
    bf16_t hb = (bf16_t)hn;
    h_bf[i] = hb;
    att_in[b * 1024 + HH + h] = hb;   // h part of next step's attention input
    xh_g [b * 1024 + HH + h] = hb;   // h part of next step's gate-GEMM input
}

__global__ __launch_bounds__(128) void k_loss_argmax(const float* __restrict__ out_logits,
                                                     int t, const int* __restrict__ target,
                                                     const bf16_t* __restrict__ emb_bf,
                                                     bf16_t* __restrict__ att_in,
                                                     bf16_t* __restrict__ comb_in,
                                                     float* __restrict__ loss_acc)
{   // one block of 128 per batch row: log-softmax loss + argmax + gather next embedding
    __shared__ float sv[128]; __shared__ int si[128]; __shared__ float ss[128];
    int b = blockIdx.x, tid = threadIdx.x;
    const float* row = out_logits + (size_t)b * LDOUT + t * VV;
    float lg = row[tid];
    sv[tid] = lg; si[tid] = tid; __syncthreads();
    for (int s = 64; s > 0; s >>= 1) {
        if (tid < s) {
            float o = sv[tid + s]; int oi = si[tid + s];
            if (o > sv[tid] || (o == sv[tid] && oi < si[tid])) { sv[tid] = o; si[tid] = oi; }
        }
        __syncthreads();
    }
    float mx = sv[0]; int best = si[0]; __syncthreads();
    ss[tid] = expf(lg - mx); __syncthreads();
    for (int s = 64; s > 0; s >>= 1) { if (tid < s) ss[tid] += ss[tid + s]; __syncthreads(); }
    if (tid == 0) {
        int y = target[b * MAXLEN + t];
        float loss = logf(ss[0]) + mx - row[y];
        atomicAdd(loss_acc, loss);
    }
    // gather next-step token embedding (x_emb for both att and comb inputs)
    for (int k = tid; k < HH; k += 128) {
        bf16_t e = emb_bf[best * HH + k];
        att_in [b * 1024 + k] = e;
        comb_in[b * 1024 + k] = e;
    }
}

__global__ void k_finalize(const float* __restrict__ loss_acc, float* __restrict__ out)
{
    if (threadIdx.x == 0) *out = *loss_acc / (float)(MAXLEN * BB);
}

// =====================================================================
extern "C" void kernel_launch(void* const* d_in, const int* in_sizes, int n_in,
                              void* d_out, int out_size, void* d_ws, size_t ws_size,
                              hipStream_t stream)
{
    (void)in_sizes; (void)n_in; (void)out_size; (void)ws_size;
    const int*   enc_out = (const int*)  d_in[0];
    const int*   target  = (const int*)  d_in[1];
    const float* emb     = (const float*)d_in[2];
    const float* att_W   = (const float*)d_in[3];
    const float* att_b   = (const float*)d_in[4];
    const float* comb_W  = (const float*)d_in[5];
    const float* comb_b  = (const float*)d_in[6];
    const float* W_ih    = (const float*)d_in[7];
    const float* W_hh    = (const float*)d_in[8];
    const float* b_ih    = (const float*)d_in[9];
    const float* b_hh    = (const float*)d_in[10];
    const float* fc_W    = (const float*)d_in[11];
    const float* fc_b    = (const float*)d_in[12];
    float* out = (float*)d_out;

    // -------- workspace layout (bytes, 256-aligned) --------
    char* ws = (char*)d_ws;
    size_t off = 0;
    auto alloc = [&](size_t bytes) { char* p = ws + off; off = (off + bytes + 255) & ~(size_t)255; return p; };
    bf16_t* emb_bf   = (bf16_t*)alloc((size_t)VV * HH * 2);
    bf16_t* attW_bf  = (bf16_t*)alloc((size_t)NPAD * 2 * HH * 2);
    bf16_t* combW_bf = (bf16_t*)alloc((size_t)HH * 2 * HH * 2);
    bf16_t* Wg_bf    = (bf16_t*)alloc((size_t)4 * HH * 2 * HH * 2);
    bf16_t* fcW_bf   = (bf16_t*)alloc((size_t)VV * HH * 2);
    float*  bsum     = (float*) alloc((size_t)4 * HH * 4);
    bf16_t* enc_bf   = (bf16_t*)alloc((size_t)BB * SS * HH * 2);   // 64 MB
    bf16_t* att_in   = (bf16_t*)alloc((size_t)BB * 1024 * 2);      // [x_emb | h]
    bf16_t* comb_in  = (bf16_t*)alloc((size_t)BB * 1024 * 2);      // [x_emb | a]
    bf16_t* xh_g     = (bf16_t*)alloc((size_t)BB * 1024 * 2);      // [x | h]
    float*  att_log  = (float*) alloc((size_t)BB * NPAD * 4);
    float*  att_wf   = (float*) alloc((size_t)BB * 256 * 4);
    float*  gates    = (float*) alloc((size_t)BB * 4 * HH * 4);
    bf16_t* h_bf     = (bf16_t*)alloc((size_t)BB * HH * 2);
    float*  c_st     = (float*) alloc((size_t)BB * HH * 4);
    float*  loss_acc = (float*) alloc(256);

    // -------- one-time setup --------
    k_cast_bf16<<<(VV * HH + 255) / 256, 256, 0, stream>>>(emb, emb_bf, VV * HH);
    k_cast_bf16<<<(HH * 2 * HH + 255) / 256, 256, 0, stream>>>(comb_W, combW_bf, HH * 2 * HH);
    k_cast_bf16<<<(VV * HH + 255) / 256, 256, 0, stream>>>(fc_W, fcW_bf, VV * HH);
    k_build_attW<<<(NPAD * 2 * HH + 255) / 256, 256, 0, stream>>>(att_W, attW_bf);
    k_build_Wg<<<(4 * HH * 2 * HH + 255) / 256, 256, 0, stream>>>(W_ih, W_hh, Wg_bf);
    k_build_bsum<<<(4 * HH + 255) / 256, 256, 0, stream>>>(b_ih, b_hh, bsum);
    k_build_encoded<<<(BB * SS * HH + 255) / 256, 256, 0, stream>>>(enc_out, emb, enc_bf);
    k_init_state<<<(BB * HH + 255) / 256, 256, 0, stream>>>(emb_bf, c_st, h_bf, att_in,
                                                            comb_in, xh_g, loss_acc);

    // -------- 300 serial decode steps --------
    for (int t = 0; t < MAXLEN; ++t) {
        // att logits [256,300] = att_in[256,1024] @ attW^T  (+att_b)
        gemm_bf16_wmma<false, true, false><<<dim3(BB / 16, 3), 256, 0, stream>>>(
            att_in, attW_bf, att_b, att_log, NPAD, nullptr, 0, 0,
            MAXLEN, NPAD / 16, 2 * HH);
        k_softmax_att<<<BB, 256, 0, stream>>>(att_log, att_wf);
        k_context<<<BB, 256, 0, stream>>>(att_wf, enc_bf, comb_in);
        // x = relu(comb_in @ comb_W^T + comb_b) -> bf16 into xh_g[:, 0:512]
        gemm_bf16_wmma<true, false, true><<<dim3(BB / 16, 4), 256, 0, stream>>>(
            comb_in, combW_bf, comb_b, nullptr, 0, xh_g, 1024, 0,
            HH, HH / 16, 2 * HH);
        // gates = xh_g @ [W_ih|W_hh]^T + (b_ih + b_hh)
        gemm_bf16_wmma<false, true, false><<<dim3(BB / 16, 16), 256, 0, stream>>>(
            xh_g, Wg_bf, bsum, gates, 4 * HH, nullptr, 0, 0,
            4 * HH, (4 * HH) / 16, 2 * HH);
        k_lstm_ew<<<(BB * HH + 255) / 256, 256, 0, stream>>>(gates, c_st, h_bf, att_in, xh_g);
        // logits -> d_out[b, t, :]
        gemm_bf16_wmma<false, true, false><<<dim3(BB / 16, 1), 256, 0, stream>>>(
            h_bf, fcW_bf, fc_b, out + t * VV, LDOUT, nullptr, 0, 0,
            VV, VV / 16, HH);
        k_loss_argmax<<<BB, 128, 0, stream>>>(out, t, target, emb_bf, att_in, comb_in, loss_acc);
    }

    k_finalize<<<1, 1, 0, stream>>>(loss_acc, out + (size_t)BB * MAXLEN * VV);
}
